// ResidualGNN_15685220565567
// MI455X (gfx1250) — compile-verified, matrix operand order checked by reading, not run on hardware
//
#include <hip/hip_runtime.h>
#include <hip/hip_bf16.h>

// ---------------------------------------------------------------------------
// CDNA5 (gfx1250) implementation: bf16 WMMA GEMMs + atomic scatter GCN.
// Wave32; each 256-thread block = 8 waves arranged 2(M) x 4(N); each wave
// computes a 16x64 output tile with four v_wmma_f32_16x16x32_bf16.
// ---------------------------------------------------------------------------

typedef __bf16 v16bf __attribute__((ext_vector_type(16)));
typedef __bf16 v8bf  __attribute__((ext_vector_type(8)));
typedef float  v8f   __attribute__((ext_vector_type(8)));

#define NN   50000
#define NE   400000
#define FN   64
#define FE   16
#define HID  256
#define KPROJ 544          // 2*HID + FE = 528, padded to multiple of 32

static __device__ __forceinline__ __bf16 f2bf(float f) {
  union { float f; unsigned u; } in; in.f = f;
  unsigned r = in.u + 0x7FFFu + ((in.u >> 16) & 1u);   // round-to-nearest-even
  union { unsigned short s; __bf16 b; } out; out.s = (unsigned short)(r >> 16);
  return out.b;
}

// Fragment load: 16 bf16 per lane as two contiguous 16B chunks
// (matches ISA 16-bit A layout: VGPR0-3 <- K=base..base+7, VGPR4-7 <- K=base+16..+23)
static __device__ __forceinline__ v16bf load_frag16(const __bf16* p) {
  v8bf lo = *(const v8bf*)(p);
  v8bf hi = *(const v8bf*)(p + 16);
  return __builtin_shufflevector(lo, hi, 0,1,2,3,4,5,6,7,8,9,10,11,12,13,14,15);
}

static __device__ __forceinline__ float wave_sum32(float v) {
#pragma unroll
  for (int m = 16; m >= 1; m >>= 1) v += __shfl_xor(v, m, 32);
  return v;
}

// ---------------------------------------------------------------------------
// Weight transpose + fp32->bf16 cast: src [K,N] row-major -> dst [N,Kp] bf16
// (pad K..Kp-1 with zeros).
// ---------------------------------------------------------------------------
__global__ void transpose_cast_kernel(const float* __restrict__ src,
                                      __bf16* __restrict__ dst,
                                      int K, int N, int Kp) {
  int idx = blockIdx.x * blockDim.x + threadIdx.x;
  if (idx >= N * Kp) return;
  int n = idx / Kp, k = idx - n * Kp;
  float v = (k < K) ? src[(size_t)k * N + n] : 0.0f;
  dst[idx] = f2bf(v);
}

// ---------------------------------------------------------------------------
// Generic GEMM: C[M,256] = act(A[M,K]_f32 @ W + bias), W given as Bt[256,K] bf16.
// A converted to bf16 while staging into LDS. K % 32 == 0.
// ---------------------------------------------------------------------------
template <int RELU>
__global__ __launch_bounds__(256)
void gemm_n256(const float* __restrict__ A, const __bf16* __restrict__ Bt,
               const float* __restrict__ bias, float* __restrict__ C,
               int M, int K) {
  __shared__ __bf16 As[32][40];                 // 32x32 tile, padded stride (80B rows)
  const int tid  = threadIdx.x;
  const int lane = tid & 31, wave = tid >> 5;
  const int waveM = wave >> 2, waveN = wave & 3;
  const int half = lane >> 4, l16 = lane & 15;
  const int rowBase = blockIdx.x * 32;

  v8f acc[4] = {v8f{}, v8f{}, v8f{}, v8f{}};

  const int srow = tid >> 3;                    // staging row (0..31)
  const int scol = (tid & 7) * 4;               // staging col (0,4,...,28)
  const int grow = rowBase + srow;

  for (int kt = 0; kt < K; kt += 32) {
    // ---- stage A tile (fp32 -> bf16) ----
    float4 f = make_float4(0.f, 0.f, 0.f, 0.f);
    if (grow < M) {
      f = *(const float4*)(A + (size_t)grow * K + kt + scol);
      if (kt + 32 < K)
        __builtin_prefetch(A + (size_t)grow * K + kt + 32 + scol, 0, 1);
    }
    As[srow][scol + 0] = f2bf(f.x); As[srow][scol + 1] = f2bf(f.y);
    As[srow][scol + 2] = f2bf(f.z); As[srow][scol + 3] = f2bf(f.w);
    __syncthreads();

    // ---- fragments + WMMA ----
    v16bf a = load_frag16(&As[waveM * 16 + l16][half * 8]);
    const __bf16* bp = Bt + (size_t)(waveN * 64 + l16) * K + kt + half * 8;
#pragma unroll
    for (int nt = 0; nt < 4; ++nt) {
      v16bf b = load_frag16(bp + (size_t)nt * 16 * K);
      acc[nt] = __builtin_amdgcn_wmma_f32_16x16x32_bf16(
          false, a, false, b, (short)0, acc[nt], false, false);
    }
    __syncthreads();
  }

  // ---- epilogue: bias (+ReLU), C/D layout: VGPR r -> M=r+half*8, lane%16 -> N ----
#pragma unroll
  for (int nt = 0; nt < 4; ++nt) {
    int col = waveN * 64 + nt * 16 + l16;
    float bv = bias[col];
#pragma unroll
    for (int r = 0; r < 8; ++r) {
      int row = rowBase + waveM * 16 + half * 8 + r;
      if (row < M) {
        float v = acc[nt][r] + bv;
        if (RELU) v = fmaxf(v, 0.f);
        C[(size_t)row * HID + col] = v;
      }
    }
  }
}

// ---------------------------------------------------------------------------
// Fused edge gather + concat + projection GEMM:
// eh[e,:] = relu( [x[src[e]] | x[dst[e]] | edge_attr[e]] @ proj_W + proj_b )
// Concat row (528, padded to 544) staged in LDS as bf16; Wt is [256,544] bf16.
// ---------------------------------------------------------------------------
__global__ __launch_bounds__(256)
void proj_gemm(const float* __restrict__ x, const long long* __restrict__ src,
               const long long* __restrict__ dst, const float* __restrict__ ea,
               const __bf16* __restrict__ Wt, const float* __restrict__ bias,
               float* __restrict__ C) {
  __shared__ __bf16 Es[32][KPROJ + 8];          // 1104B rows (16B aligned)
  const int tid  = threadIdx.x;
  const int lane = tid & 31, wave = tid >> 5;
  const int waveM = wave >> 2, waveN = wave & 3;
  const int half = lane >> 4, l16 = lane & 15;
  const int eBase = blockIdx.x * 32;            // NE % 32 == 0

  for (int idx = tid; idx < 32 * KPROJ; idx += 256) {
    int r = idx / KPROJ, c = idx - r * KPROJ;
    int e = eBase + r;
    float v = 0.f;
    if (c < HID) {
      v = x[(size_t)src[e] * HID + c];
    } else if (c < 2 * HID) {
      v = x[(size_t)dst[e] * HID + (c - HID)];
    } else if (c < 2 * HID + FE) {
      v = ea[(size_t)e * FE + (c - 2 * HID)];
    }
    Es[r][c] = f2bf(v);
  }
  __syncthreads();

  v8f acc[4] = {v8f{}, v8f{}, v8f{}, v8f{}};
  for (int kt = 0; kt < KPROJ; kt += 32) {
    v16bf a = load_frag16(&Es[waveM * 16 + l16][kt + half * 8]);
    const __bf16* bp = Wt + (size_t)(waveN * 64 + l16) * KPROJ + kt + half * 8;
#pragma unroll
    for (int nt = 0; nt < 4; ++nt) {
      v16bf b = load_frag16(bp + (size_t)nt * 16 * KPROJ);
      acc[nt] = __builtin_amdgcn_wmma_f32_16x16x32_bf16(
          false, a, false, b, (short)0, acc[nt], false, false);
    }
  }

#pragma unroll
  for (int nt = 0; nt < 4; ++nt) {
    int col = waveN * 64 + nt * 16 + l16;
    float bv = bias[col];
#pragma unroll
    for (int r = 0; r < 8; ++r) {
      int row = eBase + waveM * 16 + half * 8 + r;
      C[(size_t)row * HID + col] = fmaxf(acc[nt][r] + bv, 0.f);
    }
  }
}

// ---------------------------------------------------------------------------
// GCN helpers
// ---------------------------------------------------------------------------
__global__ void deg_count_kernel(const long long* __restrict__ dst,
                                 float* __restrict__ deg, int E) {
  int e = blockIdx.x * blockDim.x + threadIdx.x;
  if (e < E) atomicAdd(deg + dst[e], 1.0f);
}

__global__ void deg_to_dinv_kernel(float* __restrict__ deg, int n) {
  int i = blockIdx.x * blockDim.x + threadIdx.x;
  if (i < n) deg[i] = rsqrtf(deg[i] + 1.0f);    // +1 self-loop
}

// agg[dst] += h[src] * dinv[src]*dinv[dst]  (8 threads per edge, 32 cols each)
__global__ void gcn_scatter_kernel(const long long* __restrict__ src,
                                   const long long* __restrict__ dst,
                                   const float* __restrict__ h,
                                   const float* __restrict__ dinv,
                                   float* __restrict__ agg, int E) {
  int gid = blockIdx.x * blockDim.x + threadIdx.x;
  int e = gid >> 3;
  if (e >= E) return;
  int c0 = (gid & 7) * 32;
  long long s = src[e], d = dst[e];
  float coef = dinv[s] * dinv[d];
  const float* hp = h + (size_t)s * HID + c0;
  float* ap = agg + (size_t)d * HID + c0;
#pragma unroll 4
  for (int c = 0; c < 32; ++c) atomicAdd(ap + c, hp[c] * coef);
}

// x = LayerNorm(agg + h*dinv^2 + x) ; one wave per row, 8 cols/lane
__global__ __launch_bounds__(256)
void gcn_post_kernel(const float* __restrict__ agg, const float* __restrict__ h,
                     const float* __restrict__ dinv, const float* __restrict__ g,
                     const float* __restrict__ b, float* __restrict__ x, int n) {
  int row = blockIdx.x * 8 + (threadIdx.x >> 5);
  int lane = threadIdx.x & 31;
  if (row >= n) return;
  float dv = dinv[row], sl = dv * dv;
  size_t base = (size_t)row * HID + lane * 8;
  float v[8];
#pragma unroll
  for (int j = 0; j < 8; ++j) v[j] = agg[base + j] + h[base + j] * sl + x[base + j];
  float s = 0.f;
#pragma unroll
  for (int j = 0; j < 8; ++j) s += v[j];
  float mu = wave_sum32(s) * (1.0f / HID);
  float sq = 0.f;
#pragma unroll
  for (int j = 0; j < 8; ++j) { float d = v[j] - mu; sq += d * d; }
  float rs = rsqrtf(wave_sum32(sq) * (1.0f / HID) + 1e-5f);
#pragma unroll
  for (int j = 0; j < 8; ++j) {
    int col = lane * 8 + j;
    x[base + j] = (v[j] - mu) * rs * g[col] + b[col];
  }
}

// h = LayerNorm(u + h)  (classifier residual LN)
__global__ __launch_bounds__(256)
void cls_post_kernel(const float* __restrict__ u, float* __restrict__ h,
                     const float* __restrict__ g, const float* __restrict__ b, int n) {
  int row = blockIdx.x * 8 + (threadIdx.x >> 5);
  int lane = threadIdx.x & 31;
  if (row >= n) return;
  size_t base = (size_t)row * HID + lane * 8;
  float v[8];
#pragma unroll
  for (int j = 0; j < 8; ++j) v[j] = u[base + j] + h[base + j];
  float s = 0.f;
#pragma unroll
  for (int j = 0; j < 8; ++j) s += v[j];
  float mu = wave_sum32(s) * (1.0f / HID);
  float sq = 0.f;
#pragma unroll
  for (int j = 0; j < 8; ++j) { float d = v[j] - mu; sq += d * d; }
  float rs = rsqrtf(wave_sum32(sq) * (1.0f / HID) + 1e-5f);
#pragma unroll
  for (int j = 0; j < 8; ++j) {
    int col = lane * 8 + j;
    h[base + j] = (v[j] - mu) * rs * g[col] + b[col];
  }
}

// logits[e] = h[e] @ out_W[256,2] + out_b  (out_W cached in LDS)
__global__ __launch_bounds__(256)
void out_head_kernel(const float* __restrict__ h, const float* __restrict__ W,
                     const float* __restrict__ b, float* __restrict__ out, int E) {
  __shared__ float w[HID * 2];
  for (int i = threadIdx.x; i < HID * 2; i += 256) w[i] = W[i];
  __syncthreads();
  int e = blockIdx.x * blockDim.x + threadIdx.x;
  if (e >= E) return;
  float a0 = b[0], a1 = b[1];
  const float* hp = h + (size_t)e * HID;
#pragma unroll 4
  for (int k = 0; k < HID; ++k) {
    float xv = hp[k];
    a0 += xv * w[2 * k + 0];
    a1 += xv * w[2 * k + 1];
  }
  out[2 * e + 0] = a0;
  out[2 * e + 1] = a1;
}

// ---------------------------------------------------------------------------
// Host launcher
// ---------------------------------------------------------------------------
extern "C" void kernel_launch(void* const* d_in, const int* in_sizes, int n_in,
                              void* d_out, int out_size, void* d_ws, size_t ws_size,
                              hipStream_t stream) {
  (void)in_sizes; (void)n_in; (void)out_size; (void)ws_size;

  const float*     node_f  = (const float*)d_in[0];
  const long long* eidx    = (const long long*)d_in[1];   // int64 [2,E]
  const float*     ea      = (const float*)d_in[2];
  const float*     enc_W   = (const float*)d_in[3];
  const float*     enc_b   = (const float*)d_in[4];
  const float*     gnn_W   = (const float*)d_in[5];
  const float*     gnn_b   = (const float*)d_in[6];
  const float*     gln_g   = (const float*)d_in[7];
  const float*     gln_b   = (const float*)d_in[8];
  const float*     proj_W  = (const float*)d_in[9];
  const float*     proj_b  = (const float*)d_in[10];
  const float*     cls_W1  = (const float*)d_in[11];
  const float*     cls_b1  = (const float*)d_in[12];
  const float*     cls_W2  = (const float*)d_in[13];
  const float*     cls_b2  = (const float*)d_in[14];
  const float*     cln_g   = (const float*)d_in[15];
  const float*     cln_b   = (const float*)d_in[16];
  const float*     out_W   = (const float*)d_in[17];
  const float*     out_b   = (const float*)d_in[18];

  const long long* src = eidx;
  const long long* dst = eidx + NE;

  // --- workspace layout (256B aligned slabs) ---
  char* ws = (char*)d_ws;
  size_t off = 0;
  auto take = [&](size_t bytes) { char* p = ws + off; off += (bytes + 255) & ~(size_t)255; return p; };
  float*  x      = (float*)take((size_t)NN * HID * 4);
  float*  hbuf   = (float*)take((size_t)NN * HID * 4);
  float*  agg    = (float*)take((size_t)NN * HID * 4);
  float*  dinv   = (float*)take((size_t)NN * 4);
  __bf16* encWt  = (__bf16*)take((size_t)HID * FN * 2);
  __bf16* gnnWt  = (__bf16*)take((size_t)4 * HID * HID * 2);
  __bf16* projWt = (__bf16*)take((size_t)HID * KPROJ * 2);
  __bf16* c1Wt   = (__bf16*)take((size_t)2 * HID * HID * 2);
  __bf16* c2Wt   = (__bf16*)take((size_t)2 * HID * HID * 2);
  float*  eh     = (float*)take((size_t)NE * HID * 4);
  float*  et     = (float*)take((size_t)NE * HID * 4);

  // --- 1. weight transpose/cast to bf16 [N,Kp] ---
  transpose_cast_kernel<<<(HID * FN + 255) / 256, 256, 0, stream>>>(enc_W, encWt, FN, HID, FN);
  for (int i = 0; i < 4; ++i)
    transpose_cast_kernel<<<(HID * HID + 255) / 256, 256, 0, stream>>>(
        gnn_W + (size_t)i * HID * HID, gnnWt + (size_t)i * HID * HID, HID, HID, HID);
  transpose_cast_kernel<<<(HID * KPROJ + 255) / 256, 256, 0, stream>>>(
      proj_W, projWt, 2 * HID + FE, HID, KPROJ);
  for (int i = 0; i < 2; ++i) {
    transpose_cast_kernel<<<(HID * HID + 255) / 256, 256, 0, stream>>>(
        cls_W1 + (size_t)i * HID * HID, c1Wt + (size_t)i * HID * HID, HID, HID, HID);
    transpose_cast_kernel<<<(HID * HID + 255) / 256, 256, 0, stream>>>(
        cls_W2 + (size_t)i * HID * HID, c2Wt + (size_t)i * HID * HID, HID, HID, HID);
  }

  // --- 2. degrees -> dinv ---
  hipMemsetAsync(dinv, 0, (size_t)NN * 4, stream);
  deg_count_kernel<<<(NE + 255) / 256, 256, 0, stream>>>(dst, dinv, NE);
  deg_to_dinv_kernel<<<(NN + 255) / 256, 256, 0, stream>>>(dinv, NN);

  // --- 3. node encoder: x = relu(nf @ enc_W + enc_b) ---
  gemm_n256<1><<<(NN + 31) / 32, 256, 0, stream>>>(node_f, encWt, enc_b, x, NN, FN);

  // --- 4. GCN residual blocks ---
  for (int i = 0; i < 4; ++i) {
    gemm_n256<0><<<(NN + 31) / 32, 256, 0, stream>>>(
        x, gnnWt + (size_t)i * HID * HID, gnn_b + (size_t)i * HID, hbuf, NN, HID);
    hipMemsetAsync(agg, 0, (size_t)NN * HID * 4, stream);
    gcn_scatter_kernel<<<(NE * 8 + 255) / 256, 256, 0, stream>>>(src, dst, hbuf, dinv, agg, NE);
    gcn_post_kernel<<<(NN + 7) / 8, 256, 0, stream>>>(
        agg, hbuf, dinv, gln_g + (size_t)i * HID, gln_b + (size_t)i * HID, x, NN);
  }

  // --- 5. fused edge gather + projection ---
  proj_gemm<<<NE / 32, 256, 0, stream>>>(x, src, dst, ea, projWt, proj_b, eh);

  // --- 6. classifier residual blocks ---
  for (int i = 0; i < 2; ++i) {
    gemm_n256<1><<<NE / 32, 256, 0, stream>>>(
        eh, c1Wt + (size_t)i * HID * HID, cls_b1 + (size_t)i * HID, et, NE, HID);
    gemm_n256<0><<<NE / 32, 256, 0, stream>>>(          // in-place: block writes only rows it read
        et, c2Wt + (size_t)i * HID * HID, cls_b2 + (size_t)i * HID, et, NE, HID);
    cls_post_kernel<<<NE / 8, 256, 0, stream>>>(
        et, eh, cln_g + (size_t)i * HID, cln_b + (size_t)i * HID, NE);
  }

  // --- 7. output head ---
  out_head_kernel<<<(NE + 255) / 256, 256, 0, stream>>>(eh, out_W, out_b, (float*)d_out, NE);
}
